// SpatAttn_15298673508950
// MI455X (gfx1250) — matrix operand order repaired and checked by reading, not compiled, not run on hardware
//
#include <hip/hip_runtime.h>
#include <hip/hip_bf16.h>

// ---------------------------------------------------------------------------
// SpatAttn for MI455X (gfx1250, wave32, WMMA).
// Shapes: B=8, C=256, D=32, H=W=64 -> N=4096.
// Pipeline:
//   k_prep:      [wq;wk;wv] f32 -> wb (320x256) bf16   (one-time, 160 KB)
//   k_transpose: x (B,C,N) f32  -> xt (B,N,C) bf16     (LDS-tiled)
//   k_proj:      WMMA GEMM wb(320x256) @ x(256xN):
//                  qT,kT -> (B,N,D) bf16 ; v -> (B,C,N) bf16
//   k_attn:      flash attention: S = qT*k (WMMA K=32), online softmax,
//                O += P*V (WMMA), epilogue gamma*O/l + x
// ---------------------------------------------------------------------------

typedef __bf16 bf16_t;
typedef __attribute__((ext_vector_type(16))) __bf16 v16bf;
typedef __attribute__((ext_vector_type(8)))  float  v8f;

#define BATCH 8
#define CDIM  256
#define DDIM  32
#define NDIM  4096
#define WROWS 320   // 32 q + 32 k + 256 v

static __device__ __forceinline__ bf16_t f2bf(float f) {
  // round-to-nearest-even f32 -> bf16
  unsigned int u = __builtin_bit_cast(unsigned int, f);
  u += 0x7fffu + ((u >> 16) & 1u);
  unsigned short h = (unsigned short)(u >> 16);
  return __builtin_bit_cast(bf16_t, h);
}

union Frag16 { v16bf v; uint4 q[2]; };

static __device__ __forceinline__ v8f wmma_bf16(v16bf a, v16bf b, v8f c) {
  // (neg_a, A, neg_b, B, c_mod, C, reuse_a, reuse_b)
  return __builtin_amdgcn_wmma_f32_16x16x32_bf16(false, a, false, b, (short)0, c,
                                                 false, false);
}

// B-fragment (bf16 32x16, K x Ncol) from a source where K is contiguous per
// column: element (k, ncol) at colbase[ncol*ldk + k].
// Layout used here (contiguous k): lane l -> ncol = l%16, khalf = (l/16)*16, elements = K khalf..+15.
static __device__ __forceinline__ v16bf load_b_cols(const bf16_t* colbase,
                                                    size_t ldk, int lane) {
  int l16 = lane & 15;
  int h16 = (lane >> 4) << 4;
  return *(const v16bf*)(colbase + (size_t)l16 * ldk + h16);
}

// A-fragment (bf16 16x32, Mrow x K) given rowp = &src[Mrow_for_this_lane][0]
// with K contiguous. Layout used here (contiguous k): lane l -> M = l%16, kb16 = (l/16)*16,
// elements 0..15 = K kb16..kb16+15.
static __device__ __forceinline__ v16bf load_a_row(const bf16_t* rowp, int lane) {
  int kb8 = (lane >> 4) << 4;
  Frag16 f;
  f.q[0] = *(const uint4*)(rowp + kb8);
  f.q[1] = *(const uint4*)(rowp + 8 + kb8);
  return f.v;
}

// ---------------------------------------------------------------------------
// Kernel P: one-time weight conversion [wq;wk;wv] -> wb (320x256 bf16).
// ---------------------------------------------------------------------------
__global__ __launch_bounds__(256) void k_prep(const float* __restrict__ wq,
                                              const float* __restrict__ wk,
                                              const float* __restrict__ wv,
                                              bf16_t* __restrict__ wb) {
  int i = blockIdx.x * 256 + threadIdx.x;        // 0 .. 320*256-1
  int row = i >> 8;
  int col = i & 255;
  float v;
  if (row < 32)        v = wq[(size_t)row * CDIM + col];
  else if (row < 64)   v = wk[(size_t)(row - 32) * CDIM + col];
  else                 v = wv[(size_t)(row - 64) * CDIM + col];
  wb[i] = f2bf(v);
}

// ---------------------------------------------------------------------------
// Kernel 0: x (B,C,N) f32 -> xt (B,N,C) bf16, 32x32 LDS tiles.
// ---------------------------------------------------------------------------
__global__ __launch_bounds__(256) void k_transpose(const float* __restrict__ x,
                                                   bf16_t* __restrict__ xt) {
  __shared__ bf16_t tile[32][33];
  const int tx = threadIdx.x & 31;
  const int ty = threadIdx.x >> 5;          // 0..7
  const int n0 = blockIdx.x * 32;
  const int c0 = blockIdx.y * 32;
  const int b  = blockIdx.z;
  const float* xb = x + (size_t)b * CDIM * NDIM;
#pragma unroll
  for (int i = 0; i < 4; ++i) {
    int c = ty + i * 8;
    tile[c][tx] = f2bf(xb[(size_t)(c0 + c) * NDIM + n0 + tx]);
  }
  __syncthreads();
  bf16_t* xtb = xt + (size_t)b * NDIM * CDIM;
#pragma unroll
  for (int i = 0; i < 4; ++i) {
    int n = ty + i * 8;
    xtb[(size_t)(n0 + n) * CDIM + c0 + tx] = tile[tx][n];
  }
}

// ---------------------------------------------------------------------------
// Kernel 1: projections via WMMA, all-bf16 operands (no inner-loop cvt).
// Grid: BATCH * (NDIM/128) blocks, 256 threads (8 waves).
// Wave w owns n-tile n0 = chunk*128 + w*16; preloads all 8 B-frags (K=256)
// from xt once, then sweeps 20 output row tiles of wb = [wq;wk;wv].
// ---------------------------------------------------------------------------
__global__ __launch_bounds__(256) void k_proj(
    const bf16_t* __restrict__ xt, const bf16_t* __restrict__ wb,
    const float* __restrict__ bq, const float* __restrict__ bk,
    const float* __restrict__ bv,
    bf16_t* __restrict__ qT, bf16_t* __restrict__ kT, bf16_t* __restrict__ vB) {
  const int wave = threadIdx.x >> 5;
  const int lane = threadIdx.x & 31;
  const int l16  = lane & 15;
  const int half = lane >> 4;
  const int b    = blockIdx.x >> 5;         // 32 n-chunks per batch
  const int nch  = blockIdx.x & 31;
  const int n0   = nch * 128 + wave * 16;

  // Preload 8 B-fragments: column n = n0 + l16, K runs kc*32 + half*16 .. +15.
  const bf16_t* colp =
      xt + ((size_t)b * NDIM + n0 + l16) * CDIM + (half << 4);
  v16bf xf[8];
#pragma unroll
  for (int kc = 0; kc < 8; ++kc) xf[kc] = *(const v16bf*)(colp + kc * 32);

  // One 16-row output tile: A-frags are straight b128 loads from bf16 wb.
  auto gemm_tile = [&](int row0) -> v8f {
    v8f acc = {};
    const bf16_t* wrow = wb + (size_t)(row0 + l16) * CDIM;
#pragma unroll
    for (int kc = 0; kc < 8; ++kc)
      acc = wmma_bf16(load_a_row(wrow + kc * 32, lane), xf[kc], acc);
    return acc;
  };

  // D-frag layout: lane l, reg r -> row = r + 8*(l/16), col = l%16.
  auto store_qk = [&](v8f acc, const float* bias, int d0, bf16_t* dst) {
#pragma unroll
    for (int r = 0; r < 8; ++r) {
      int d = d0 + r + 8 * half;
      dst[((size_t)b * NDIM + n0 + l16) * DDIM + d] = f2bf(acc[r] + bias[d]);
    }
  };
  auto store_v = [&](v8f acc, int c0) {
#pragma unroll
    for (int r = 0; r < 8; ++r) {
      int c = c0 + r + 8 * half;
      vB[((size_t)b * CDIM + c) * NDIM + n0 + l16] = f2bf(acc[r] + bv[c]);
    }
  };

  store_qk(gemm_tile(0),  bq, 0,  qT);
  store_qk(gemm_tile(16), bq, 16, qT);
  store_qk(gemm_tile(32), bk, 0,  kT);
  store_qk(gemm_tile(48), bk, 16, kT);
#pragma unroll 1
  for (int t = 0; t < 16; ++t) store_v(gemm_tile(64 + t * 16), t * 16);
}

// ---------------------------------------------------------------------------
// Kernel 2: flash attention.
// Grid: BATCH * (NDIM/16) blocks, 128 threads (4 waves).
// Each wave: full S tile + softmax, private 64-col slice of C for P*V.
// ---------------------------------------------------------------------------
__global__ __launch_bounds__(128) void k_attn(
    const float* __restrict__ xres,
    const bf16_t* __restrict__ qT, const bf16_t* __restrict__ kT,
    const bf16_t* __restrict__ vB, const float* __restrict__ gamma,
    float* __restrict__ out) {
  __shared__ __align__(16) bf16_t pshm[4][16 * 32];   // per-wave P buffer

  const int wave = threadIdx.x >> 5;
  const int lane = threadIdx.x & 31;
  const int l16  = lane & 15;
  const int half = lane >> 4;
  const int b    = blockIdx.x >> 8;         // 256 query tiles per batch
  const int n0   = (blockIdx.x & 255) * 16;
  const int c0w  = wave * 64;

  // A-fragment of q^T: M = query row, K = d (0..31).
  const v16bf aq =
      load_a_row(qT + ((size_t)b * NDIM + n0 + l16) * DDIM, lane);

  float rmax[8], rsum[8];
  v8f oacc[4];
#pragma unroll
  for (int r = 0; r < 8; ++r) { rmax[r] = -3.0e38f; rsum[r] = 0.0f; }
#pragma unroll
  for (int t = 0; t < 4; ++t) oacc[t] = (v8f){};

  const bf16_t* kbase = kT + (size_t)b * NDIM * DDIM;
  const bf16_t* vbase = vB + (size_t)b * CDIM * NDIM;
  bf16_t* pw = pshm[wave];

#pragma unroll 1
  for (int m0 = 0; m0 < NDIM; m0 += 32) {
    // S = q^T * k for 32 keys: two WMMAs, K-dim = D = 32 exactly.
    v8f z = {};
    v8f s0 = wmma_bf16(aq, load_b_cols(kbase + (size_t)m0 * DDIM, DDIM, lane), z);
    v8f s1 = wmma_bf16(aq, load_b_cols(kbase + (size_t)(m0 + 16) * DDIM, DDIM, lane), z);
    if (m0 + 32 < NDIM)
      __builtin_prefetch(kbase + (size_t)(m0 + 32) * DDIM + lane * 8, 0, 1);

    // Online softmax. Row of element r is (r + 8*half); its 16 columns live
    // across lanes {group*16 + 0..15} -> width-16 xor reductions.
    float nscale[8];
#pragma unroll
    for (int r = 0; r < 8; ++r) {
      float mx = fmaxf(s0[r], s1[r]);
#pragma unroll
      for (int off = 8; off > 0; off >>= 1) mx = fmaxf(mx, __shfl_xor(mx, off, 16));
      float nm = fmaxf(rmax[r], mx);
      float sc = __expf(rmax[r] - nm);
      rmax[r] = nm;
      float p0 = __expf(s0[r] - nm);
      float p1 = __expf(s1[r] - nm);
      s0[r] = p0; s1[r] = p1;
      float ps = p0 + p1;
#pragma unroll
      for (int off = 8; off > 0; off >>= 1) ps += __shfl_xor(ps, off, 16);
      rsum[r] = rsum[r] * sc + ps;
      nscale[r] = sc;
    }
#pragma unroll
    for (int t = 0; t < 4; ++t)
#pragma unroll
      for (int r = 0; r < 8; ++r) oacc[t][r] *= nscale[r];

    // Re-lay P out as an A-fragment via LDS (C/D layout -> A layout).
#pragma unroll
    for (int r = 0; r < 8; ++r) {
      int row = r + 8 * half;
      pw[row * 32 + l16]      = f2bf(s0[r]);
      pw[row * 32 + 16 + l16] = f2bf(s1[r]);
    }
    __syncthreads();   // uniform; orders DS stores vs DS loads
    v16bf ap = load_a_row(pw + l16 * 32, lane);

    // O += P * V over this wave's 64 channels (4 c-tiles, K=32 per WMMA).
#pragma unroll
    for (int t = 0; t < 4; ++t) {
      const bf16_t* vcol = vbase + (size_t)(c0w + t * 16) * NDIM + m0;
      oacc[t] = wmma_bf16(ap, load_b_cols(vcol, NDIM, lane), oacc[t]);
    }
    __syncthreads();
  }

  // Epilogue: out = gamma * O / rowsum + x.
  const float g = gamma[0];
  float rinv[8];
#pragma unroll
  for (int r = 0; r < 8; ++r) rinv[r] = g / rsum[r];
#pragma unroll
  for (int t = 0; t < 4; ++t) {
    int c = c0w + t * 16 + l16;
    size_t base = ((size_t)b * CDIM + c) * NDIM + n0 + 8 * half;
#pragma unroll
    for (int r = 0; r < 8; ++r)
      out[base + r] = oacc[t][r] * rinv[r] + xres[base + r];
  }
}

// ---------------------------------------------------------------------------
extern "C" void kernel_launch(void* const* d_in, const int* in_sizes, int n_in,
                              void* d_out, int out_size, void* d_ws, size_t ws_size,
                              hipStream_t stream) {
  (void)in_sizes; (void)n_in; (void)out_size; (void)ws_size;
  const float* x     = (const float*)d_in[0];
  const float* wq    = (const float*)d_in[1];
  const float* bq    = (const float*)d_in[2];
  const float* wk    = (const float*)d_in[3];
  const float* bk    = (const float*)d_in[4];
  const float* wv    = (const float*)d_in[5];
  const float* bv    = (const float*)d_in[6];
  const float* gamma = (const float*)d_in[7];
  float* out = (float*)d_out;

  // Workspace carve-up (~36.2 MiB total, L2-resident on MI455X).
  char* ws = (char*)d_ws;
  const size_t XT_BYTES = (size_t)BATCH * NDIM * CDIM * sizeof(bf16_t); // 16 MiB
  const size_t QB_BYTES = (size_t)BATCH * NDIM * DDIM * sizeof(bf16_t); //  2 MiB
  bf16_t* xt = (bf16_t*)(ws);
  bf16_t* qT = (bf16_t*)(ws + XT_BYTES);
  bf16_t* kT = (bf16_t*)(ws + XT_BYTES + QB_BYTES);
  bf16_t* vB = (bf16_t*)(ws + XT_BYTES + 2 * QB_BYTES);
  bf16_t* wbuf = (bf16_t*)(ws + 2 * XT_BYTES + 2 * QB_BYTES);           // 160 KiB

  k_prep<<<(WROWS * CDIM) / 256, 256, 0, stream>>>(wq, wk, wv, wbuf);
  k_transpose<<<dim3(NDIM / 32, CDIM / 32, BATCH), 256, 0, stream>>>(x, xt);
  k_proj<<<BATCH * (NDIM / 128), 256, 0, stream>>>(xt, wbuf, bq, bk, bv,
                                                   qT, kT, vB);
  k_attn<<<BATCH * (NDIM / 16), 128, 0, stream>>>(x, qT, kT, vB, gamma, out);
}
